// MultiHeadAttention_90924457656763
// MI455X (gfx1250) — matrix operand order repaired and checked by reading, not compiled, not run on hardware
//
#include <hip/hip_runtime.h>
#include <hip/hip_bf16.h>

#define BB   2
#define SS   2048
#define DD   1024
#define HH   16
#define HDIM 64
// GEMM: M = BB*SS = 4096, N = HH*HDIM = 1024, K = DD = 1024

typedef __bf16 bf16;
typedef __attribute__((ext_vector_type(16))) __bf16 v16bf;
typedef __attribute__((ext_vector_type(8)))  __bf16 v8bf;
typedef __attribute__((ext_vector_type(8)))  float  v8f;
typedef __attribute__((ext_vector_type(4)))  unsigned int u32x4;
typedef __attribute__((ext_vector_type(8)))  int i32x8;
typedef __attribute__((ext_vector_type(4)))  int i32x4;

#if defined(__has_builtin)
#if __has_builtin(__builtin_amdgcn_tensor_load_to_lds) && \
    __has_builtin(__builtin_amdgcn_s_wait_tensorcnt)
#define USE_TDM 1
#endif
#endif
#ifndef USE_TDM
#define USE_TDM 0
#endif

static __device__ __forceinline__ v16bf cat8(v8bf lo, v8bf hi) {
  v16bf r;
#pragma unroll
  for (int i = 0; i < 8; ++i) { r[i] = lo[i]; r[i + 8] = hi[i]; }
  return r;
}

static __device__ __forceinline__ v8f wmma_bf16(v16bf a, v16bf b, v8f c) {
  // D = A(16x32 bf16) * B(32x16 bf16) + C(16x16 f32)
  return __builtin_amdgcn_wmma_f32_16x16x32_bf16(
      false, a, false, b, (short)0, c, false, false);
}

static __device__ __forceinline__ float rowmax16(float x) {
#pragma unroll
  for (int m = 1; m < 16; m <<= 1) x = fmaxf(x, __shfl_xor(x, m, 32));
  return x;
}
static __device__ __forceinline__ float rowsum16(float x) {
#pragma unroll
  for (int m = 1; m < 16; m <<= 1) x += __shfl_xor(x, m, 32);
  return x;
}

#if USE_TDM
static __device__ __forceinline__ unsigned lds_byte_off(const void* p) {
  // generic -> LDS(as3) pointer; as3 pointers are 32-bit LDS byte offsets
  return (unsigned)(size_t)(__attribute__((address_space(3))) const void*)p;
}

// Issue a TDM DMA of a 64-row x 32-col bf16 tile (rows stride 1024 elems)
// from global into LDS (rows packed contiguously, 64B apart).
static __device__ __forceinline__ void tdm_stage_b(unsigned lds_addr,
                                                   const bf16* gsrc) {
  const unsigned long long ga = (unsigned long long)(size_t)gsrc;
  u32x4 g0;
  g0[0] = 1u;                                   // count=1 (valid user D#)
  g0[1] = lds_addr;                             // lds_addr (bytes)
  g0[2] = (unsigned)(ga & 0xffffffffu);         // global_addr[31:0]
  g0[3] = (unsigned)((ga >> 32) & 0x01ffffffu)  // global_addr[56:32]
          | 0x80000000u;                        // type=2 ("image")
  i32x8 g1;
  g1[0] = 1 << 16;                              // data_size=1 -> 2 bytes
  g1[1] = (int)(1024u << 16);                   // tensor_dim0 = 1024
  g1[2] = (int)(1024u << 16);                   // tensor_dim1 = 1024
  g1[3] = (int)(32u << 16);                     // tile_dim0   = 32 (k)
  g1[4] = 64;                                   // tile_dim1   = 64 (n rows)
  g1[5] = 1024;                                 // tensor_dim0_stride = 1024
  g1[6] = 0;
  g1[7] = 0;
  i32x4 z4; z4[0] = z4[1] = z4[2] = z4[3] = 0;  // 2D tensor: groups 2/3 unused
  i32x8 z8;
#pragma unroll
  for (int i = 0; i < 8; ++i) z8[i] = 0;
  // 6-arg form (clang-23 / therock-10.0 headers): extra int32x8 before cpol
  __builtin_amdgcn_tensor_load_to_lds(g0, g1, z4, z4, z8, 0);
}
#endif

// ---------------------------------------------------------------------------
// Kernel 0: f32 -> bf16 conversion of X, and transposed bf16 weights Wt[n][k]
// ---------------------------------------------------------------------------
__global__ void prep_kernel(const float* __restrict__ X,
                            const float* __restrict__ Wq,
                            const float* __restrict__ Wk,
                            const float* __restrict__ Wv,
                            bf16* __restrict__ Xbf,   // [4096][1024]
                            bf16* __restrict__ Wt) {  // [3][1024(N)][1024(K)]
  const int NX = BB * SS * DD;
  const int NW = DD * HH * HDIM;
  const int stride = gridDim.x * blockDim.x;
  for (int i = blockIdx.x * blockDim.x + threadIdx.x; i < NX; i += stride)
    Xbf[i] = (bf16)X[i];
  for (int i = blockIdx.x * blockDim.x + threadIdx.x; i < NW; i += stride) {
    const int n = i >> 10, k = i & 1023;
    const int src = k * 1024 + n;   // W is [K][N] row-major
    Wt[i]          = (bf16)Wq[src];
    Wt[NW + i]     = (bf16)Wk[src];
    Wt[2 * NW + i] = (bf16)Wv[src];
  }
}

// ---------------------------------------------------------------------------
// Kernel 1: QKV projection GEMM. 8 waves/block, wave -> 16x64 strip.
//   B tile (64x32 weights, shared by all 8 waves) staged in LDS, double
//   buffered; staged by the Tensor Data Mover when available.
//   Q,K stored [BH][S][64] bf16 ; V stored transposed [BH][64][S] bf16
// ---------------------------------------------------------------------------
__global__ __launch_bounds__(256) void qkv_gemm_kernel(
    const bf16* __restrict__ Xbf,  // [4096][1024]
    const bf16* __restrict__ Wt,   // [3][1024][1024] (transposed weights)
    bf16* __restrict__ Qbf,        // [32][2048][64]
    bf16* __restrict__ Kbf,        // [32][2048][64]
    bf16* __restrict__ Vt) {       // [32][64][2048]
  __shared__ __align__(16) bf16 Blds[2][64][32];   // 2 x 4KB B-tile buffers

  const int w    = blockIdx.z;           // 0=Q 1=K 2=V
  const int tid  = threadIdx.x;
  const int lane = tid & 31;
  const int wave = tid >> 5;
  const int lm   = lane & 15;
  const int half = lane >> 4;
  const int m0   = blockIdx.x * 128 + wave * 16;
  const int h    = blockIdx.y;           // 64-wide N tile == one head
  const int n0   = h * 64;
  const bf16* Wb    = Wt + (size_t)w * (DD * HH * HDIM);
  const bf16* tile0 = Wb + (size_t)n0 * DD;        // row n0, col 0

#if !USE_TDM
  const int cn = tid >> 2;               // staging: row 0..63
  const int ck = (tid & 3) * 8;          // staging: col segment
#endif

  // ---- stage first B tile (kb=0) into buffer 0 ----
#if USE_TDM
  if (wave == 0) {
    tdm_stage_b(lds_byte_off(&Blds[0][0][0]), tile0);
    __builtin_amdgcn_s_wait_tensorcnt(0);
  }
  asm volatile("" ::: "memory");         // LDS written by DMA, not visible stores
#else
  *(v8bf*)&Blds[0][cn][ck] = *(const v8bf*)(tile0 + (size_t)cn * DD + ck);
#endif
  __syncthreads();

  v8f vz = {};
  v8f acc[4] = {vz, vz, vz, vz};

  const bf16* arow = Xbf + (size_t)(m0 + lm) * DD;
  v16bf a = cat8(*(const v8bf*)(arow + 16 * half),
                 *(const v8bf*)(arow + 8 + 16 * half));

  int buf = 0;
  for (int kb = 0; kb < DD; kb += 32) {
    const int nkb = kb + 32;
    const bool more = (nkb < DD);

    // ---- issue staging of next B tile into buf^1 (overlaps WMMAs) ----
#if !USE_TDM
    v8bf stage_reg;
#endif
    if (more) {
#if USE_TDM
      if (wave == 0)
        tdm_stage_b(lds_byte_off(&Blds[buf ^ 1][0][0]), tile0 + nkb);
#else
      stage_reg = *(const v8bf*)(tile0 + (size_t)cn * DD + nkb + ck);
#endif
    }

    // ---- prefetch next A fragment into registers ----
    v16bf an = a;
    if (more)
      an = cat8(*(const v8bf*)(arow + nkb + 16 * half),
                *(const v8bf*)(arow + nkb + 8 + 16 * half));
    if (nkb + 64 < DD) __builtin_prefetch(arow + nkb + 64, 0, 3);

    // ---- read all 4 B fragments first, then 4 back-to-back WMMAs ----
    v16bf bfr[4];
#pragma unroll
    for (int nb = 0; nb < 4; ++nb)
      bfr[nb] = *(const v16bf*)&Blds[buf][nb * 16 + lm][16 * half];
#pragma unroll
    for (int nb = 0; nb < 4; ++nb)
      acc[nb] = wmma_bf16(a, bfr[nb], acc[nb]);

    // ---- commit staging, flip buffers ----
    if (more) {
#if USE_TDM
      if (wave == 0) __builtin_amdgcn_s_wait_tensorcnt(0);
#else
      *(v8bf*)&Blds[buf ^ 1][cn][ck] = stage_reg;
#endif
      a = an;
#if USE_TDM
      asm volatile("" ::: "memory");
#endif
      __syncthreads();
    }
    buf ^= 1;
  }

  const int b  = m0 >> 11;               // batch (tile never crosses S)
  const int bh = b * HH + h;
#pragma unroll
  for (int nb = 0; nb < 4; ++nb) {
    const int d = nb * 16 + lm;
#pragma unroll
    for (int j = 0; j < 8; ++j) {
      const int m = m0 + j + 8 * half;   // C layout: M = j + 8*half
      const int s = m & (SS - 1);
      const bf16 val = (bf16)acc[nb][j];
      if (w == 0)      Qbf[((size_t)bh * SS + s) * HDIM + d] = val;
      else if (w == 1) Kbf[((size_t)bh * SS + s) * HDIM + d] = val;
      else             Vt[((size_t)bh * HDIM + d) * SS + s]  = val;
    }
  }
}

// ---------------------------------------------------------------------------
// Kernel 2: flash attention. 4 waves/block; wave = 16 query rows of one head.
// K fragments double-buffered; V loads issued before softmax so the VALU
// softmax section hides global latency.
// ---------------------------------------------------------------------------
__global__ __launch_bounds__(128) void attn_kernel(
    const bf16* __restrict__ Qbf,  // [32][2048][64]
    const bf16* __restrict__ Kbf,  // [32][2048][64]
    const bf16* __restrict__ Vt,   // [32][64][2048]
    float* __restrict__ Out) {     // [2][2048][1024]
  __shared__ __align__(16) bf16 Plds[4][16][32];   // per-wave P staging

  const int bh   = blockIdx.x;          // b*16 + h
  const int lane = threadIdx.x & 31;
  const int wave = threadIdx.x >> 5;
  const int lm   = lane & 15;
  const int half = lane >> 4;
  const int s0   = blockIdx.y * 64 + wave * 16;

  const bf16* Qh = Qbf + (size_t)bh * SS * HDIM;
  const bf16* Kh = Kbf + (size_t)bh * SS * HDIM;
  const bf16* Vh = Vt  + (size_t)bh * HDIM * SS;

  // Q A-fragments for k(d) = 0..31 and 32..63, loaded once
  const bf16* qrow = Qh + (size_t)(s0 + lm) * HDIM;
  v16bf qa0 = cat8(*(const v8bf*)(qrow + 16 * half),
                   *(const v8bf*)(qrow + 8 + 16 * half));
  v16bf qa1 = cat8(*(const v8bf*)(qrow + 32 + 16 * half),
                   *(const v8bf*)(qrow + 40 + 16 * half));

  v8f vz = {};
  v8f o[4] = {vz, vz, vz, vz};
  float mrow[8], lrow[8];
#pragma unroll
  for (int j = 0; j < 8; ++j) { mrow[j] = -1e30f; lrow[j] = 0.f; }

  const float sc = 0.125f;  // 1/sqrt(HD)

  // preload K fragments for first 32-key block
  v16bf kf[4];
  {
    const bf16* kr0 = Kh + (size_t)lm * HDIM;
    const bf16* kr1 = Kh + (size_t)(16 + lm) * HDIM;
    kf[0] = *(const v16bf*)(kr0 + 16 * half);
    kf[1] = *(const v16bf*)(kr0 + 32 + 16 * half);
    kf[2] = *(const v16bf*)(kr1 + 16 * half);
    kf[3] = *(const v16bf*)(kr1 + 32 + 16 * half);
  }

  for (int t0 = 0; t0 < SS; t0 += 32) {
    // ---- issue V fragment loads early (consumed after softmax) ----
    v16bf vf[4];
#pragma unroll
    for (int nb = 0; nb < 4; ++nb)
      vf[nb] = *(const v16bf*)(Vh + (size_t)(nb * 16 + lm) * SS + t0 + 16 * half);

    // ---- scores for 32 keys: two 16-key C fragments, 2 WMMAs each ----
    v8f sf0 = vz, sf1 = vz;
    sf0 = wmma_bf16(qa0, kf[0], sf0);
    sf0 = wmma_bf16(qa1, kf[1], sf0);
    sf1 = wmma_bf16(qa0, kf[2], sf1);
    sf1 = wmma_bf16(qa1, kf[3], sf1);

    // ---- issue next block's K fragment loads (hidden by softmax VALU) ----
    const bool more = (t0 + 32 < SS);
    v16bf kfn[4];
    if (more) {
      const bf16* kr0 = Kh + (size_t)(t0 + 32 + lm) * HDIM;
      const bf16* kr1 = Kh + (size_t)(t0 + 48 + lm) * HDIM;
      kfn[0] = *(const v16bf*)(kr0 + 16 * half);
      kfn[1] = *(const v16bf*)(kr0 + 32 + 16 * half);
      kfn[2] = *(const v16bf*)(kr1 + 16 * half);
      kfn[3] = *(const v16bf*)(kr1 + 32 + 16 * half);
    }

    // ---- online softmax (one m-update per 32-key block) ----
#pragma unroll
    for (int j = 0; j < 8; ++j) {
      const float x0 = sf0[j] * sc;
      const float x1 = sf1[j] * sc;
      const float bm = rowmax16(fmaxf(x0, x1));
      const float mnew = fmaxf(mrow[j], bm);
      const float alpha = __expf(mrow[j] - mnew);
      mrow[j] = mnew;
      const float p0 = __expf(x0 - mnew);
      const float p1 = __expf(x1 - mnew);
      lrow[j] = lrow[j] * alpha + rowsum16(p0 + p1);
#pragma unroll
      for (int nb = 0; nb < 4; ++nb) o[nb][j] *= alpha;
      const int r = j + 8 * half;       // C layout row
      Plds[wave][r][lm]      = (bf16)p0;
      Plds[wave][r][16 + lm] = (bf16)p1;
    }

    // ---- P (C layout) -> A layout through LDS, then P*V: 4 WMMAs ----
    const bf16* prow = &Plds[wave][lm][0];
    v16bf pa = cat8(*(const v8bf*)(prow + 16 * half),
                    *(const v8bf*)(prow + 8 + 16 * half));
#pragma unroll
    for (int nb = 0; nb < 4; ++nb)
      o[nb] = wmma_bf16(pa, vf[nb], o[nb]);

    if (more) {
#pragma unroll
      for (int i = 0; i < 4; ++i) kf[i] = kfn[i];
    }
  }

  // ---- normalize and write [B,S,H*HD] f32 ----
  const int b = bh >> 4, h = bh & 15;
#pragma unroll
  for (int j = 0; j < 8; ++j) {
    const float inv = 1.0f / lrow[j];
    const int s = s0 + j + 8 * half;
    float* orow = Out + ((size_t)b * SS + s) * (HH * HDIM) + h * HDIM;
#pragma unroll
    for (int nb = 0; nb < 4; ++nb)
      orow[nb * 16 + lm] = o[nb][j] * inv;
  }
}

// ---------------------------------------------------------------------------
extern "C" void kernel_launch(void* const* d_in, const int* in_sizes, int n_in,
                              void* d_out, int out_size, void* d_ws, size_t ws_size,
                              hipStream_t stream) {
  (void)in_sizes; (void)n_in; (void)out_size; (void)ws_size;
  const float* X  = (const float*)d_in[0];
  const float* Wq = (const float*)d_in[1];
  const float* Wk = (const float*)d_in[2];
  const float* Wv = (const float*)d_in[3];
  float* Out = (float*)d_out;

  char* ws = (char*)d_ws;
  const size_t NX = (size_t)BB * SS * DD;        // 4,194,304
  const size_t NW = (size_t)DD * HH * HDIM;      // 1,048,576
  const size_t NQ = (size_t)BB * HH * SS * HDIM; // 4,194,304
  bf16* Xbf = (bf16*)ws;                            // 8 MB
  bf16* Wt  = (bf16*)(ws + NX * 2);                 // 6 MB
  bf16* Qbf = (bf16*)(ws + NX * 2 + 3 * NW * 2);    // 8 MB
  bf16* Kbf = Qbf + NQ;                             // 8 MB
  bf16* Vt  = Kbf + NQ;                             // 8 MB  (total 38 MB)

  prep_kernel<<<1024, 256, 0, stream>>>(X, Wq, Wk, Wv, Xbf, Wt);
  qkv_gemm_kernel<<<dim3(32, 16, 3), 256, 0, stream>>>(Xbf, Wt, Qbf, Kbf, Vt);
  attn_kernel<<<dim3(32, 32), 128, 0, stream>>>(Qbf, Kbf, Vt, Out);
}
